// MultiHeadAttention_19696720019913
// MI455X (gfx1250) — matrix operand order repaired and checked by reading, not compiled, not run on hardware
//
#include <hip/hip_runtime.h>
#include <hip/hip_bf16.h>

#define D_MODELC 1024
#define HEADSC   16
#define DKC      64
#define BATCHC   4
#define SEQC     2048
#define MROWSC   (BATCHC * SEQC)   // 8192

typedef __attribute__((ext_vector_type(16))) __bf16 v16bf;
typedef __attribute__((ext_vector_type(8)))  float  v8f;

union Frag {
    v16bf v;
    unsigned short u[16];
    uint4 q[2];
};

__device__ __forceinline__ unsigned short f2bf(float f) {
    unsigned u = __builtin_bit_cast(unsigned, f);
    u += 0x7FFFu + ((u >> 16) & 1u);            // round-to-nearest-even
    return (unsigned short)(u >> 16);
}

// A-matrix (16x32 bf16) per-lane element->K map (ISA 7.12.2):
// lanes 0-15: e<8 -> K=e,   e>=8 -> K=e+8 ; lanes 16-31: same + 8
__device__ __forceinline__ int mapA(int lane, int e) {
    return ((lane >> 1) & 8) + (e & 7) + ((e & 8) << 1);
}

__device__ __forceinline__ v8f wmma_bf16(const Frag& a, const Frag& b, v8f c) {
    return __builtin_amdgcn_wmma_f32_16x16x32_bf16(
        false, a.v, false, b.v, (short)0, c, false, false);
}

// ---------------------------------------------------------------- fp32 -> bf16 bulk convert
__global__ void cvt_bf16_kernel(const float* __restrict__ src,
                                unsigned short* __restrict__ dst, int n) {
    int i = blockIdx.x * blockDim.x + threadIdx.x;
    if (i < n) dst[i] = f2bf(src[i]);
}

// ---------------------------------------------------------------- Q/K/V projection
// y = X @ W^T + bias ; X already bf16 ; B-frag B[kk][n] = W[n][kk] (row-major -> contiguous)
// 32x32 output tile per wave: 2 A-frags x 2 B-frags -> 4 WMMAs per 32-deep K step.
// MODE 0: store [b, h, s, dk]   (Q, K)
// MODE 1: store [b, h, dk, s]   (V transposed for contiguous P*V B-fragments)
template <int MODE>
__global__ __launch_bounds__(256) void proj_kernel(
    const unsigned short* __restrict__ Xb, const unsigned short* __restrict__ Wbf,
    const float* __restrict__ bias, unsigned short* __restrict__ out)
{
    const int lane = threadIdx.x & 31;
    const int wave = threadIdx.x >> 5;
    const int tile = blockIdx.x * 8 + wave;        // 8192 tiles (256 x 32)
    const int tn = tile & 31;                      // N/32 = 32
    const int tm = tile >> 5;                      // M/32 = 256
    const int nl = lane & 15;
    const int kh = lane & 16;
    const int Mb = kh >> 1;

    const unsigned short* xrow0 = Xb + (size_t)(tm * 32 + nl) * D_MODELC;
    const unsigned short* xrow1 = xrow0 + 16 * D_MODELC;
    const unsigned short* wrow0 = Wbf + (size_t)(tn * 32 + nl) * D_MODELC;
    const unsigned short* wrow1 = wrow0 + 16 * D_MODELC;

    v8f c00 = {}, c01 = {}, c10 = {}, c11 = {};
    for (int k = 0; k < D_MODELC; k += 32) {
        Frag a0, a1, b0, b1;
        a0.q[0] = *(const uint4*)(xrow0 + k + Mb);
        a0.q[1] = *(const uint4*)(xrow0 + k + Mb + 16);
        a1.q[0] = *(const uint4*)(xrow1 + k + Mb);
        a1.q[1] = *(const uint4*)(xrow1 + k + Mb + 16);
        b0.q[0] = *(const uint4*)(wrow0 + k + Mb);
        b0.q[1] = *(const uint4*)(wrow0 + k + Mb + 16);
        b1.q[0] = *(const uint4*)(wrow1 + k + Mb);
        b1.q[1] = *(const uint4*)(wrow1 + k + Mb + 16);
        c00 = wmma_bf16(a0, b0, c00);
        c01 = wmma_bf16(a0, b1, c01);
        c10 = wmma_bf16(a1, b0, c10);
        c11 = wmma_bf16(a1, b1, c11);
    }

    v8f* accs[4] = { &c00, &c01, &c10, &c11 };
#pragma unroll
    for (int i = 0; i < 2; ++i) {
#pragma unroll
        for (int j = 0; j < 2; ++j) {
            const v8f& c = *accs[i * 2 + j];
            const int n = tn * 32 + j * 16 + nl;
            const float bn = bias[n];
            const int h = n >> 6, dk = n & 63;
#pragma unroll
            for (int r = 0; r < 8; ++r) {
                int row = tm * 32 + i * 16 + Mb + r;
                int bi = row >> 11, s = row & (SEQC - 1);
                unsigned short val = f2bf(c[r] + bn);
                size_t idx;
                if (MODE == 0) idx = ((size_t)(bi * HEADSC + h) * SEQC + s) * DKC + dk;
                else           idx = ((size_t)(bi * HEADSC + h) * DKC + dk) * SEQC + s;
                out[idx] = val;
            }
        }
    }
}

// ---------------------------------------------------------------- attention block body
// MASK=true only for the final (diagonal-crossing) 32-key block of each query tile.
template <bool MASK>
__device__ __forceinline__ void attn_block(
    const int k0, const int q0,
    const int lane, const int nl, const int kh, const int Mb,
    const unsigned short* __restrict__ Kbh, const unsigned short* __restrict__ Vth,
    const Frag& aq0, const Frag& aq1, unsigned short* __restrict__ Pl,
    v8f acc[4], float mrun[8], float lrun[8])
{
    // ---- issue ALL fragment loads up front (independent registers -> one big clause)
    Frag bk00, bk01, bk10, bk11;     // K: tile0/tile1 x dk-chunk0/1
    {
        const unsigned short* kr0 = Kbh + (size_t)(k0 + nl) * DKC;
        const unsigned short* kr1 = Kbh + (size_t)(k0 + 16 + nl) * DKC;
        bk00.q[0] = *(const uint4*)(kr0 + Mb);
        bk00.q[1] = *(const uint4*)(kr0 + Mb + 16);
        bk01.q[0] = *(const uint4*)(kr0 + 32 + Mb);
        bk01.q[1] = *(const uint4*)(kr0 + 32 + Mb + 16);
        bk10.q[0] = *(const uint4*)(kr1 + Mb);
        bk10.q[1] = *(const uint4*)(kr1 + Mb + 16);
        bk11.q[0] = *(const uint4*)(kr1 + 32 + Mb);
        bk11.q[1] = *(const uint4*)(kr1 + 32 + Mb + 16);
    }
    Frag bv0, bv1, bv2, bv3;          // V: four 32x16 column tiles (latency hides under softmax)
    {
        const unsigned short* vb = Vth + k0 + Mb;
        bv0.q[0] = *(const uint4*)(vb + (size_t)(nl)      * SEQC);
        bv0.q[1] = *(const uint4*)(vb + (size_t)(nl)      * SEQC + 16);
        bv1.q[0] = *(const uint4*)(vb + (size_t)(16 + nl) * SEQC);
        bv1.q[1] = *(const uint4*)(vb + (size_t)(16 + nl) * SEQC + 16);
        bv2.q[0] = *(const uint4*)(vb + (size_t)(32 + nl) * SEQC);
        bv2.q[1] = *(const uint4*)(vb + (size_t)(32 + nl) * SEQC + 16);
        bv3.q[0] = *(const uint4*)(vb + (size_t)(48 + nl) * SEQC);
        bv3.q[1] = *(const uint4*)(vb + (size_t)(48 + nl) * SEQC + 16);
    }

    // ---- scores: two 16x16 tiles, K-dim 64 = 2 WMMAs each
    v8f s0 = {}, s1 = {};
    s0 = wmma_bf16(aq0, bk00, s0);
    s0 = wmma_bf16(aq1, bk01, s0);
    s1 = wmma_bf16(aq0, bk10, s1);
    s1 = wmma_bf16(aq1, bk11, s1);

    // ---- online softmax (row = q0+Mb+r ; columns spread over 16-lane groups)
    float p0[8], p1[8], fac[8];
#pragma unroll
    for (int r = 0; r < 8; ++r) {
        float x0 = s0[r] * 0.125f;
        float x1 = s1[r] * 0.125f;
        if (MASK) {
            const int q = q0 + Mb + r;
            if (k0 + nl > q)      x0 = -1e30f;
            if (k0 + 16 + nl > q) x1 = -1e30f;
        }
        float mx = fmaxf(x0, x1);
        mx = fmaxf(mx, __shfl_xor(mx, 1, 32));
        mx = fmaxf(mx, __shfl_xor(mx, 2, 32));
        mx = fmaxf(mx, __shfl_xor(mx, 4, 32));
        mx = fmaxf(mx, __shfl_xor(mx, 8, 32));
        const float mn = fmaxf(mrun[r], mx);
        fac[r] = __expf(mrun[r] - mn);
        p0[r]  = __expf(x0 - mn);
        p1[r]  = __expf(x1 - mn);
        float ps = p0[r] + p1[r];
        ps += __shfl_xor(ps, 1, 32);
        ps += __shfl_xor(ps, 2, 32);
        ps += __shfl_xor(ps, 4, 32);
        ps += __shfl_xor(ps, 8, 32);
        lrun[r] = lrun[r] * fac[r] + ps;
        mrun[r] = mn;
    }
#pragma unroll
    for (int r = 0; r < 8; ++r) {
        acc[0][r] *= fac[r]; acc[1][r] *= fac[r];
        acc[2][r] *= fac[r]; acc[3][r] *= fac[r];
    }

    // ---- C-layout P -> LDS -> A-layout P fragment (wave-private, no barrier)
#pragma unroll
    for (int r = 0; r < 8; ++r) {
        const int Mr = Mb + r;
        Pl[Mr * 32 + nl]      = f2bf(p0[r]);
        Pl[Mr * 32 + 16 + nl] = f2bf(p1[r]);
    }
    asm volatile("s_wait_dscnt 0" ::: "memory");
    Frag pf;
    {
        const unsigned short* pr = Pl + nl * 32;
#pragma unroll
        for (int e = 0; e < 16; ++e) pf.u[e] = pr[mapA(lane, e)];
    }

    // ---- acc += P(16x32) * V(32x64)
    acc[0] = wmma_bf16(pf, bv0, acc[0]);
    acc[1] = wmma_bf16(pf, bv1, acc[1]);
    acc[2] = wmma_bf16(pf, bv2, acc[2]);
    acc[3] = wmma_bf16(pf, bv3, acc[3]);
}

// ---------------------------------------------------------------- causal flash attention
// one wave per (b,h, 16-query tile); 32-key blocks; online softmax
__global__ __launch_bounds__(128) void attn_kernel(
    const unsigned short* __restrict__ Qb, const unsigned short* __restrict__ Kb,
    const unsigned short* __restrict__ Vt, unsigned short* __restrict__ ctx)
{
    __shared__ unsigned short ldsP[4][16 * 32];   // per-wave P-tile staging
    const int lane = threadIdx.x & 31;
    const int wave = threadIdx.x >> 5;
    unsigned short* Pl = ldsP[wave];

    const int w  = blockIdx.x * 4 + wave;          // 8192 waves
    const int qt = w & 127;                        // SEQ/16
    const int bh = w >> 7;                         // 0..63
    const int q0 = qt * 16;

    const unsigned short* Kbh = Kb + (size_t)bh * SEQC * DKC;
    const unsigned short* Vth = Vt + (size_t)bh * DKC * SEQC;
    const int nl = lane & 15;
    const int kh = lane & 16;
    const int Mb = kh >> 1;

    // Q A-fragments: dk 0..31 and 32..63 (two contiguous 16B runs per lane)
    Frag aq0, aq1;
    {
        const unsigned short* qrow = Qb + (size_t)bh * SEQC * DKC + (size_t)(q0 + nl) * DKC;
        aq0.q[0] = *(const uint4*)(qrow + Mb);
        aq0.q[1] = *(const uint4*)(qrow + Mb + 16);
        aq1.q[0] = *(const uint4*)(qrow + 32 + Mb);
        aq1.q[1] = *(const uint4*)(qrow + 32 + Mb + 16);
    }

    v8f acc[4] = {{}, {}, {}, {}};
    float mrun[8], lrun[8];
#pragma unroll
    for (int r = 0; r < 8; ++r) { mrun[r] = -1e30f; lrun[r] = 0.f; }

    const int nkb = (q0 + 47) >> 5;   // causal: only blocks with keys <= q0+15
    // steady state: blocks fully below the diagonal -> no masking
    for (int kb = 0; kb + 1 < nkb; ++kb) {
        const int k0 = kb * 32;
        __builtin_prefetch(Kbh + (size_t)(k0 + 32 + nl) * DKC, 0, 1);     // global_prefetch_b8
        __builtin_prefetch(Vth + (size_t)nl * SEQC + k0 + 32, 0, 1);
        attn_block<false>(k0, q0, lane, nl, kh, Mb, Kbh, Vth, aq0, aq1, Pl, acc, mrun, lrun);
    }
    // final block crosses the diagonal -> masked
    attn_block<true>((nkb - 1) * 32, q0, lane, nl, kh, Mb, Kbh, Vth, aq0, aq1, Pl, acc, mrun, lrun);

    // ---- normalize + store context as [b, s, h*64+dk] bf16 (row-major for out-proj)
    const int bi = bh >> 4, h = bh & 15;
#pragma unroll
    for (int r = 0; r < 8; ++r) {
        const float inv = 1.f / lrun[r];
        const int s = q0 + Mb + r;
        unsigned short* crow = ctx + (size_t)(bi * SEQC + s) * D_MODELC + h * 64;
#pragma unroll
        for (int t = 0; t < 4; ++t) crow[t * 16 + nl] = f2bf(acc[t][r] * inv);
    }
}

// ---------------------------------------------------------------- output projection (fp32 out)
// 32x32 tile per wave, same fragment scheme as proj_kernel
__global__ __launch_bounds__(256) void oproj_kernel(
    const unsigned short* __restrict__ A, const unsigned short* __restrict__ Wbf,
    const float* __restrict__ bias, float* __restrict__ out)
{
    const int lane = threadIdx.x & 31;
    const int wave = threadIdx.x >> 5;
    const int tile = blockIdx.x * 8 + wave;
    const int tn = tile & 31;
    const int tm = tile >> 5;
    const int nl = lane & 15;
    const int kh = lane & 16;
    const int Mb = kh >> 1;

    const unsigned short* arow0 = A + (size_t)(tm * 32 + nl) * D_MODELC;
    const unsigned short* arow1 = arow0 + 16 * D_MODELC;
    const unsigned short* wrow0 = Wbf + (size_t)(tn * 32 + nl) * D_MODELC;
    const unsigned short* wrow1 = wrow0 + 16 * D_MODELC;

    v8f c00 = {}, c01 = {}, c10 = {}, c11 = {};
    for (int k = 0; k < D_MODELC; k += 32) {
        Frag a0, a1, b0, b1;
        a0.q[0] = *(const uint4*)(arow0 + k + Mb);
        a0.q[1] = *(const uint4*)(arow0 + k + Mb + 16);
        a1.q[0] = *(const uint4*)(arow1 + k + Mb);
        a1.q[1] = *(const uint4*)(arow1 + k + Mb + 16);
        b0.q[0] = *(const uint4*)(wrow0 + k + Mb);
        b0.q[1] = *(const uint4*)(wrow0 + k + Mb + 16);
        b1.q[0] = *(const uint4*)(wrow1 + k + Mb);
        b1.q[1] = *(const uint4*)(wrow1 + k + Mb + 16);
        c00 = wmma_bf16(a0, b0, c00);
        c01 = wmma_bf16(a0, b1, c01);
        c10 = wmma_bf16(a1, b0, c10);
        c11 = wmma_bf16(a1, b1, c11);
    }

    v8f* accs[4] = { &c00, &c01, &c10, &c11 };
#pragma unroll
    for (int i = 0; i < 2; ++i) {
#pragma unroll
        for (int j = 0; j < 2; ++j) {
            const v8f& c = *accs[i * 2 + j];
            const int n = tn * 32 + j * 16 + nl;
            const float bn = bias[n];
#pragma unroll
            for (int r = 0; r < 8; ++r) {
                int row = tm * 32 + i * 16 + Mb + r;
                out[(size_t)row * D_MODELC + n] = c[r] + bn;
            }
        }
    }
}

// ---------------------------------------------------------------- launch
extern "C" void kernel_launch(void* const* d_in, const int* in_sizes, int n_in,
                              void* d_out, int out_size, void* d_ws, size_t ws_size,
                              hipStream_t stream)
{
    const float* q  = (const float*)d_in[0];
    const float* k  = (const float*)d_in[1];
    const float* v  = (const float*)d_in[2];
    // d_in[3] = causal mask (implicit in kernel)
    const float* Wq = (const float*)d_in[4];
    const float* bq = (const float*)d_in[5];
    const float* Wk = (const float*)d_in[6];
    const float* bk = (const float*)d_in[7];
    const float* Wv = (const float*)d_in[8];
    const float* bv = (const float*)d_in[9];
    const float* Wo = (const float*)d_in[10];
    const float* bo = (const float*)d_in[11];
    float* out = (float*)d_out;

    char* ws = (char*)d_ws;
    const size_t W_BYTES = (size_t)D_MODELC * D_MODELC * 2;            // 2 MB
    const size_t T_BYTES = (size_t)BATCHC * HEADSC * SEQC * DKC * 2;   // 16 MB
    unsigned short* Wq_bf = (unsigned short*)(ws);
    unsigned short* Wk_bf = (unsigned short*)(ws + W_BYTES);
    unsigned short* Wv_bf = (unsigned short*)(ws + 2 * W_BYTES);
    unsigned short* Wo_bf = (unsigned short*)(ws + 3 * W_BYTES);
    unsigned short* Xq_bf = (unsigned short*)(ws + 4 * W_BYTES);                 // 16 MB
    unsigned short* Xk_bf = (unsigned short*)(ws + 4 * W_BYTES + T_BYTES);
    unsigned short* Xv_bf = (unsigned short*)(ws + 4 * W_BYTES + 2 * T_BYTES);
    unsigned short* Qb    = (unsigned short*)(ws + 4 * W_BYTES + 3 * T_BYTES);
    unsigned short* Kb    = (unsigned short*)(ws + 4 * W_BYTES + 4 * T_BYTES);
    unsigned short* Vt    = (unsigned short*)(ws + 4 * W_BYTES + 5 * T_BYTES);
    unsigned short* ctx   = Xq_bf;   // Xq dead after Q projection -> alias (total ws = 104 MB)

    const int nW = D_MODELC * D_MODELC;
    cvt_bf16_kernel<<<nW / 256, 256, 0, stream>>>(Wq, Wq_bf, nW);
    cvt_bf16_kernel<<<nW / 256, 256, 0, stream>>>(Wk, Wk_bf, nW);
    cvt_bf16_kernel<<<nW / 256, 256, 0, stream>>>(Wv, Wv_bf, nW);
    cvt_bf16_kernel<<<nW / 256, 256, 0, stream>>>(Wo, Wo_bf, nW);

    const int nX = MROWSC * D_MODELC;
    cvt_bf16_kernel<<<nX / 256, 256, 0, stream>>>(q, Xq_bf, nX);
    cvt_bf16_kernel<<<nX / 256, 256, 0, stream>>>(k, Xk_bf, nX);
    cvt_bf16_kernel<<<nX / 256, 256, 0, stream>>>(v, Xv_bf, nX);

    const int proj_blocks = (MROWSC / 32) * (D_MODELC / 32) / 8;  // 1024
    proj_kernel<0><<<proj_blocks, 256, 0, stream>>>(Xq_bf, Wq_bf, bq, Qb);
    proj_kernel<0><<<proj_blocks, 256, 0, stream>>>(Xk_bf, Wk_bf, bk, Kb);
    proj_kernel<1><<<proj_blocks, 256, 0, stream>>>(Xv_bf, Wv_bf, bv, Vt);

    attn_kernel<<<BATCHC * HEADSC * (SEQC / 16) / 4, 128, 0, stream>>>(Qb, Kb, Vt, ctx);

    oproj_kernel<<<proj_blocks, 256, 0, stream>>>(ctx, Wo_bf, bo, out);
}
